// TFTest_76304388980779
// MI455X (gfx1250) — matrix operand order repaired and checked by reading, not compiled, not run on hardware
//
#include <hip/hip_runtime.h>
#include <hip/hip_bf16.h>
#include <cstdint>
#include <cstddef>

// Problem constants (match reference)
#define H_DIM   256
#define NSTATE  32      // NH
#define L_DIM   8192
#define B_DIM   16
#define BL      (B_DIM * L_DIM)      // 131072 columns of the output GEMM
#define HN      (H_DIM * NSTATE)     // 8192 (h,n) pairs

typedef __bf16    v16bf __attribute__((ext_vector_type(16)));
typedef float     v8f   __attribute__((ext_vector_type(8)));
typedef unsigned  v4u   __attribute__((ext_vector_type(4)));

union Frag16 { v4u q[2]; v16bf v; };          // 16 bf16 = 32 B (trivially constructible)
union RowU   { v4u q; unsigned short s[8]; }; // 8 bf16 row chunk for register transpose

// Uniform lane broadcast (lane index is a compile-time constant in unrolled loops):
// lowers to v_readlane_b32 -> SGPR, no LDS/bpermute traffic.
__device__ __forceinline__ float lane_bcast(float v, int l) {
  return __uint_as_float(__builtin_amdgcn_readlane(__float_as_uint(v), (unsigned)l));
}

// Wave-local LDS store->load visibility: DS ops from one wave complete in order;
// s_wait_dscnt 0 guarantees all prior DS ops are done.  The "memory" clobber is
// the compiler-side fence.  (Buffer is wave-private, so no cross-wave barrier.)
__device__ __forceinline__ void wave_lds_sync() {
  asm volatile("s_wait_dscnt 0x0" ::: "memory");
}

// ---------------------------------------------------------------------------
// K1: ZOH discretization.  per (h,n): w = exp(dt*A),  Cd = C * expm1(dt*A)/A
// ---------------------------------------------------------------------------
__global__ __launch_bounds__(256) void k_params(
    const float* __restrict__ log_dt, const float* __restrict__ A_real,
    const float* __restrict__ A_imag, const float* __restrict__ C_real,
    const float* __restrict__ C_imag,
    float* __restrict__ wr_o, float* __restrict__ wi_o,
    float* __restrict__ c0r_o, float* __restrict__ c0i_o,
    float* __restrict__ c1r_o, float* __restrict__ c1i_o) {
  int idx = blockIdx.x * blockDim.x + threadIdx.x;   // h*NSTATE + n
  if (idx >= HN) return;
  int h = idx >> 5;
  float dt = __expf(log_dt[h]);
  float Ar = -__expf(A_real[idx]);
  float Ai = A_imag[idx];
  float dr = Ar * dt, di = Ai * dt;
  float er = __expf(dr);
  float wr = er * __cosf(di), wi = er * __sinf(di);
  // expm1(dtA) = (wr-1, wi);  f = expm1(dtA) / A  (complex divide)
  float emr = wr - 1.0f, emi = wi;
  float inv = 1.0f / (Ar * Ar + Ai * Ai);
  float fr = (emr * Ar + emi * Ai) * inv;
  float fi = (emi * Ar - emr * Ai) * inv;
  float C0r = C_real[idx],      C0i = C_imag[idx];
  float C1r = C_real[HN + idx], C1i = C_imag[HN + idx];
  wr_o[idx] = wr;  wi_o[idx] = wi;
  c0r_o[idx] = C0r * fr - C0i * fi;  c0i_o[idx] = C0r * fi + C0i * fr;
  c1r_o[idx] = C1r * fr - C1i * fi;  c1i_o[idx] = C1r * fi + C1i * fr;
}

// ---------------------------------------------------------------------------
// K2: W_out f32 -> bf16 (row-major 512x256, stays L2-resident for the GEMM)
// ---------------------------------------------------------------------------
__global__ __launch_bounds__(256) void k_wb(const float* __restrict__ W,
                                            __hip_bfloat16* __restrict__ Wb) {
  int i = blockIdx.x * blockDim.x + threadIdx.x;
  if (i < 2 * H_DIM * H_DIM) Wb[i] = __float2bfloat16(W[i]);
}

// ---------------------------------------------------------------------------
// K3: transpose x (L,B,H) -> u (B,H,L), tiled 32x32 through LDS
// ---------------------------------------------------------------------------
__global__ __launch_bounds__(256) void k_transpose(const float* __restrict__ x,
                                                   float* __restrict__ u) {
  __shared__ float tile[32][33];
  int l0 = blockIdx.x * 32, h0 = blockIdx.y * 32, b = blockIdx.z;
  int tx = threadIdx.x, ty = threadIdx.y;
#pragma unroll
  for (int r = 0; r < 4; ++r) {
    int l = l0 + ty + r * 8;
    tile[ty + r * 8][tx] = x[((size_t)l * B_DIM + b) * H_DIM + h0 + tx];
  }
  __syncthreads();
#pragma unroll
  for (int r = 0; r < 4; ++r) {
    int hh = h0 + ty + r * 8;
    u[((size_t)(b * H_DIM + hh)) * L_DIM + l0 + tx] = tile[tx][ty + r * 8];
  }
}

// ---------------------------------------------------------------------------
// K4: bidirectional diagonal-SSM scan + skip + GELU, output bf16.
// One wave per (b,h); lane = state index n.  Per-step output reduction over
// the 32 states via a 32x32 LDS transpose every 32 steps (wave-private buffer,
// synced with s_wait_dscnt only -- waves drift freely for latency hiding).
//   forward : s_l = w s_{l-1} + u_l             y_f[l] = 2 Re(C0 s_l)
//   backward: r_l = w r_{l+1} + u_{l+1}         y_b[l] = 2 Re(C1 r_l)
//   y = y_f + y_b + D*u ; yg = gelu_tanh(y) -> bf16 at yg[(h*B+b)*L + l]
// ---------------------------------------------------------------------------
__global__ __launch_bounds__(256) void k_scan(
    const float* __restrict__ u,
    const float* __restrict__ wr_a, const float* __restrict__ wi_a,
    const float* __restrict__ c0r_a, const float* __restrict__ c0i_a,
    const float* __restrict__ c1r_a, const float* __restrict__ c1i_a,
    const float* __restrict__ Dvec,
    float* __restrict__ ytmp, __hip_bfloat16* __restrict__ yg) {
  __shared__ float pbuf[8][32][36];    // per-wave 32x32 transpose buffer (padded)
  int wave = threadIdx.x >> 5, lane = threadIdx.x & 31;
  int wid = blockIdx.x * 8 + wave;     // 0..4095
  int b = wid >> 8, h = wid & (H_DIM - 1);
  int pidx = h * NSTATE + lane;
  float wr = wr_a[pidx], wi = wi_a[pidx];
  float c0r = 2.0f * c0r_a[pidx], c0i = 2.0f * c0i_a[pidx];
  float c1r = 2.0f * c1r_a[pidx], c1i = 2.0f * c1i_a[pidx];
  float Dh = Dvec[h];
  const float* uptr = u + (size_t)(b * H_DIM + h) * L_DIM;
  float* yt = ytmp + (size_t)(b * H_DIM + h) * L_DIM;
  __hip_bfloat16* yo = yg + (size_t)(h * B_DIM + b) * L_DIM;
  float (*pb)[36] = pbuf[wave];

  // ---- forward scan ----
  float sr = 0.0f, si = 0.0f;
  for (int blk = 0; blk < L_DIM / 32; ++blk) {
    float uv = uptr[blk * 32 + lane];                     // coalesced 128B
    if (blk + 1 < L_DIM / 32)
      __builtin_prefetch(uptr + (blk + 1) * 32 + lane, 0, 1);
#pragma unroll
    for (int t = 0; t < 32; ++t) {
      float ut = lane_bcast(uv, t);                       // v_readlane broadcast
      float nr = fmaf(wr, sr, fmaf(-wi, si, ut));
      float ni = fmaf(wr, si, wi * sr);
      sr = nr; si = ni;
      pb[t][lane] = fmaf(c0r, sr, -c0i * si);             // per-state partial
    }
    wave_lds_sync();                                      // stores visible wave-wide
    float acc = 0.0f;
    const float4* row = (const float4*)pb[lane];          // time = blk*32+lane
#pragma unroll
    for (int n = 0; n < 8; ++n) { float4 v = row[n]; acc += v.x + v.y + v.z + v.w; }
    yt[blk * 32 + lane] = acc;                            // coalesced store
    wave_lds_sync();                                      // reads done before reuse
  }

  // ---- backward (exclusive reverse) scan, fuse skip + GELU -> bf16 ----
  float rr = 0.0f, ri = 0.0f;
  for (int blk = L_DIM / 32 - 1; blk >= 0; --blk) {
    float uv = uptr[blk * 32 + lane];
    if (blk > 0)
      __builtin_prefetch(uptr + (blk - 1) * 32 + lane, 0, 1);
#pragma unroll
    for (int t = 31; t >= 0; --t) {
      float ut = lane_bcast(uv, t);
      pb[t][lane] = fmaf(c1r, rr, -c1i * ri);             // y_b[l] uses r_l (excl.)
      float nr = fmaf(wr, rr, fmaf(-wi, ri, ut));         // r_{l-1} = w r_l + u_l
      float ni = fmaf(wr, ri, wi * rr);
      rr = nr; ri = ni;
    }
    wave_lds_sync();
    float acc = 0.0f;
    const float4* row = (const float4*)pb[lane];
#pragma unroll
    for (int n = 0; n < 8; ++n) { float4 v = row[n]; acc += v.x + v.y + v.z + v.w; }
    float y = yt[blk * 32 + lane] + acc + Dh * uv;        // uv == u at l=blk*32+lane
    float y3 = y * y * y;
    float g = 0.5f * y * (1.0f + tanhf(0.7978845608028654f * fmaf(0.044715f, y3, y)));
    yo[blk * 32 + lane] = __float2bfloat16(g);
    wave_lds_sync();
  }
}

// ---------------------------------------------------------------------------
// K5: z = Wb(512x256 bf16) x yg(256 x BL bf16) via v_wmma_f32_16x16x32_bf16,
//     fused bias + GLU:  out[b, o, l] = (z_o + b_o) * sigmoid(z_{o+256} + b_{o+256})
// Block: 256 thr (8 waves), 32 output columns.  Wave w owns M-tiles
// {2w, 2w+1} and their GLU partners {2w+16, 2w+17}; 2 N-tiles of 16.
// 8 K-steps x 8 tiles = 64 WMMA per wave.
// ---------------------------------------------------------------------------
__global__ __launch_bounds__(256) void k_gemm(
    const __hip_bfloat16* __restrict__ Wb,
    const __hip_bfloat16* __restrict__ yg,
    const float* __restrict__ b_out,
    float* __restrict__ out) {
  __shared__ __hip_bfloat16 ylds[32][264];   // [col][h], pitch 264 halves
  int colbase = blockIdx.x * 32;
  int t = threadIdx.x;

  // Stage yg tile (256 h x 32 cols) into LDS transposed to [col][h]:
  // thread owns a 4h x 8col sub-tile -> 4 global b128 loads, register
  // transpose into h-quads, 8 ds_store_b64.
  {
    int hg = t >> 2, cg = t & 3;
    int h0 = hg * 4, c0 = cg * 8;
    const __hip_bfloat16* ysrc = yg + colbase + c0;
    RowU r[4];
#pragma unroll
    for (int rr2 = 0; rr2 < 4; ++rr2)
      r[rr2].q = *(const v4u*)(ysrc + (size_t)(h0 + rr2) * BL);
#pragma unroll
    for (int cc = 0; cc < 8; ++cc) {
      unsigned lo = (unsigned)r[0].s[cc] | ((unsigned)r[1].s[cc] << 16);
      unsigned hi = (unsigned)r[2].s[cc] | ((unsigned)r[3].s[cc] << 16);
      unsigned long long q = (unsigned long long)lo | ((unsigned long long)hi << 32);
      *(unsigned long long*)&ylds[c0 + cc][h0] = q;   // 8B aligned (528|8, h0%4==0)
    }
  }
  __syncthreads();

  int wave = t >> 5, lane = t & 31;
  int lanelo = lane & 15;
  int hiHalf = (lane >> 4) & 1;

  v8f acc[2][2][2] = {};   // [g: value/gate half][j: m-subtile][nt: n-tile]

#pragma unroll
  for (int kk = 0; kk < 8; ++kk) {
    // A fragments: 16x32 bf16; lane holds row M=lanelo, K pairs per ISA layout
    int kA = kk * 32 + (hiHalf ? 8 : 0);
    Frag16 a[2][2];
#pragma unroll
    for (int g = 0; g < 2; ++g)
#pragma unroll
      for (int j = 0; j < 2; ++j) {
        int row = (2 * wave + j + g * 16) * 16 + lanelo;        // 0..511
        const __hip_bfloat16* base = Wb + (size_t)row * H_DIM + kA;
        a[g][j].q[0] = *(const v4u*)(base);        // K = kA .. kA+7
        a[g][j].q[1] = *(const v4u*)(base + 16);   // K = kA+16 .. kA+23
      }
    // B fragments: 32x16 bf16; lane holds col N=lanelo, K pairs per ISA layout (same as A)
    int kB = kk * 32 + (hiHalf ? 8 : 0);
    Frag16 bf[2];
#pragma unroll
    for (int nt = 0; nt < 2; ++nt) {
      const __hip_bfloat16* lb = &ylds[nt * 16 + lanelo][kB];   // 16B-aligned
      bf[nt].q[0] = *(const v4u*)(lb);
      bf[nt].q[1] = *(const v4u*)(lb + 16);
    }
#pragma unroll
    for (int g = 0; g < 2; ++g)
#pragma unroll
      for (int j = 0; j < 2; ++j)
#pragma unroll
        for (int nt = 0; nt < 2; ++nt)
          acc[g][j][nt] = __builtin_amdgcn_wmma_f32_16x16x32_bf16(
              false, a[g][j].v, false, bf[nt].v,
              (short)0, acc[g][j][nt], false, false);
  }

  // Epilogue: bias + GLU, store (B,H,L) f32. D-frag: elem v -> M=v+8*hiHalf.
#pragma unroll
  for (int j = 0; j < 2; ++j)
#pragma unroll
    for (int nt = 0; nt < 2; ++nt) {
      int col = colbase + nt * 16 + lanelo;
      int bb = col >> 13;              // / L_DIM
      int ll = col & (L_DIM - 1);
#pragma unroll
      for (int v = 0; v < 8; ++v) {
        int row = (2 * wave + j) * 16 + v + (hiHalf ? 8 : 0);   // 0..255
        float zA = acc[0][j][nt][v] + b_out[row];
        float zB = acc[1][j][nt][v] + b_out[row + H_DIM];
        float r = zA / (1.0f + __expf(-zB));
        out[((size_t)bb * H_DIM + row) * L_DIM + ll] = r;
      }
    }
}

// ---------------------------------------------------------------------------
// Launch.  Workspace layout (bytes):
//   [0, 196608)          : 6 x 8192 f32 param arrays (w, Cd0, Cd1)
//   [196608, 458752)     : Wb bf16 512x256
//   [458752, +134MB)     : u f32 (B,H,L)
//   [.., +134MB)         : ytmp f32 (forward scan partial)
//   [.., +67MB)          : yg bf16 [h*B + b][l]   (GEMM B matrix)
// Total ~336 MB.
// ---------------------------------------------------------------------------
extern "C" void kernel_launch(void* const* d_in, const int* in_sizes, int n_in,
                              void* d_out, int out_size, void* d_ws, size_t ws_size,
                              hipStream_t stream) {
  (void)in_sizes; (void)n_in; (void)out_size; (void)ws_size;
  const float* x      = (const float*)d_in[0];
  const float* log_dt = (const float*)d_in[1];
  const float* A_real = (const float*)d_in[2];
  const float* A_imag = (const float*)d_in[3];
  const float* C_real = (const float*)d_in[4];
  const float* C_imag = (const float*)d_in[5];
  const float* Dvec   = (const float*)d_in[6];
  const float* W_out  = (const float*)d_in[7];
  const float* b_out  = (const float*)d_in[8];
  float* out = (float*)d_out;

  char* ws = (char*)d_ws;
  float* p_wr  = (float*)(ws + 0 * (size_t)HN * 4);
  float* p_wi  = (float*)(ws + 1 * (size_t)HN * 4);
  float* p_c0r = (float*)(ws + 2 * (size_t)HN * 4);
  float* p_c0i = (float*)(ws + 3 * (size_t)HN * 4);
  float* p_c1r = (float*)(ws + 4 * (size_t)HN * 4);
  float* p_c1i = (float*)(ws + 5 * (size_t)HN * 4);
  __hip_bfloat16* Wb = (__hip_bfloat16*)(ws + 196608);
  float* u    = (float*)(ws + 458752);
  float* ytmp = (float*)(ws + 458752 + 134217728ULL);
  __hip_bfloat16* yg = (__hip_bfloat16*)(ws + 458752 + 2ULL * 134217728ULL);

  k_params<<<HN / 256, 256, 0, stream>>>(log_dt, A_real, A_imag, C_real, C_imag,
                                         p_wr, p_wi, p_c0r, p_c0i, p_c1r, p_c1i);
  k_wb<<<(2 * H_DIM * H_DIM) / 256, 256, 0, stream>>>(W_out, Wb);

  dim3 gt(L_DIM / 32, H_DIM / 32, B_DIM), bt(32, 8);
  k_transpose<<<gt, bt, 0, stream>>>(x, u);

  k_scan<<<(B_DIM * H_DIM) / 8, 256, 0, stream>>>(u, p_wr, p_wi, p_c0r, p_c0i,
                                                  p_c1r, p_c1i, Dvec, ytmp, yg);

  k_gemm<<<BL / 32, 256, 0, stream>>>(Wb, yg, b_out, out);
}
